// MultiHeadAttention_40750649704715
// MI455X (gfx1250) — matrix operand order repaired and checked by reading, not compiled, not run on hardware
//
#include <hip/hip_runtime.h>

#define EMBED    1024
#define HEADS    16
#define HEAD_DIM 64

typedef __attribute__((ext_vector_type(16))) _Float16 v16h;
typedef __attribute__((ext_vector_type(8)))  _Float16 v8h;
typedef __attribute__((ext_vector_type(8)))  float    v8f;
typedef __attribute__((ext_vector_type(8)))  int      v8i;

static __device__ __forceinline__ v8f wmma_f16(v16h a, v16h b, v8f c) {
  // D = A(16x32 f16) x B(32x16 f16) + C(16x16 f32)
  return __builtin_amdgcn_wmma_f32_16x16x32_f16(
      /*neg_a=*/false, a, /*neg_b=*/false, b,
      /*c_mod=*/(short)0, c, /*reuse_a=*/false, /*reuse_b=*/false);
}

// ---- fragment loaders -------------------------------------------------------
// A 16x32 f16 layout: lanes 0-15 (hf=0): K = 0..15; lanes 16-31: K = 16..31
static __device__ __forceinline__ v16h fragA_h(const _Float16* row, int kb, int hf) {
  const v8h lo = *(const v8h*)(row + kb + hf * 16);
  const v8h hi = *(const v8h*)(row + kb + hf * 16 + 8);
  v16h r;
#pragma unroll
  for (int i = 0; i < 8; ++i) { r[i] = lo[i]; r[i + 8] = hi[i]; }
  return r;
}
// B 32x16 f16 layout: lanes 0-15: K = 0..15; lanes 16-31: K = 16..31 (column = lane&15)
static __device__ __forceinline__ v16h fragB_h(const _Float16* row, int kb, int hf) {
  const v8h lo = *(const v8h*)(row + kb + hf * 16);
  const v8h hi = *(const v8h*)(row + kb + hf * 16 + 8);
  v16h r;
#pragma unroll
  for (int i = 0; i < 8; ++i) { r[i] = lo[i]; r[i + 8] = hi[i]; }
  return r;
}
// Same patterns but converting from f32 memory on the fly
static __device__ __forceinline__ v16h fragA_f(const float* row, int kb, int hf) {
  v16h r;
#pragma unroll
  for (int i = 0; i < 8; ++i) {
    r[i]     = (_Float16)row[kb + hf * 16 + i];
    r[i + 8] = (_Float16)row[kb + hf * 16 + 8 + i];
  }
  return r;
}
static __device__ __forceinline__ v16h fragB_f(const float* row, int kb, int hf) {
  v16h r;
#pragma unroll
  for (int i = 0; i < 16; ++i) r[i] = (_Float16)row[kb + hf * 16 + i];
  return r;
}

// ---- CDNA5 async global->LDS copy (16B per lane, tracked by ASYNCcnt) -------
static __device__ __forceinline__ void async_cp16(const _Float16* g, _Float16* l) {
  const unsigned           ldsOff = (unsigned)(uintptr_t)l;   // low 32 bits = LDS byte offset
  const unsigned long long gaddr  = (unsigned long long)(uintptr_t)g;
  asm volatile("global_load_async_to_lds_b128 %0, %1, off"
               :: "v"(ldsOff), "v"(gaddr) : "memory");
}
static __device__ __forceinline__ void wait_async0() {
  asm volatile("s_wait_asynccnt 0" ::: "memory");
}

// Build P^T B-fragment (32 keys x 16 query cols) from two C-layout score tiles.
// pw: elems 0..7 = p of first 16-key tile (this half's 8 rows), 8..15 = second tile.
static __device__ __forceinline__ v16h make_bp(v16h pw, int hf) {
  const v8i own = __builtin_bit_cast(v8i, pw);
  v8i oth;
#pragma unroll
  for (int i = 0; i < 8; ++i) oth[i] = __shfl_xor(own[i], 16, 32);
  v8i bpi;
#pragma unroll
  for (int i = 0; i < 4; ++i) {
    bpi[i]     = hf ? oth[i + 4] : own[i];      // keys {0..7 | 16..23}
    bpi[i + 4] = hf ? own[i + 4] : oth[i];      // keys {8..15 | 24..31}
  }
  return __builtin_bit_cast(v16h, bpi);
}

// ---- kernel 1: per-head projection  out = X @ W^T  (f32 -> f16) -------------
// X: [N,S,H*64] f32, W: [64,64] f32.
// transpose==0: Out[((n*H+h)*S + s)*64 + e]   (row-major per head, for Q/K)
// transpose==1: Out[((n*H+h)*64 + e)*S + s]   (V stored transposed: [N,H,D,S])
__global__ void __launch_bounds__(128) proj_kernel(const float* __restrict__ X,
                                                   const float* __restrict__ W,
                                                   _Float16* __restrict__ Out,
                                                   int S, int transpose) {
  const int lane = threadIdx.x & 31;
  const int wave = threadIdx.x >> 5;
  const int r  = lane & 15;
  const int hf = lane >> 4;
  const int h  = blockIdx.y;
  const int tok0 = (blockIdx.x * 4 + wave) * 16;
  const int n  = tok0 / S;
  const int s0 = tok0 - n * S;

  const float* xrow = X + ((size_t)tok0 + r) * EMBED + h * HEAD_DIM;
  const v16h a0 = fragA_f(xrow, 0, hf);
  const v16h a1 = fragA_f(xrow, 32, hf);

  v8f c[4];
#pragma unroll
  for (int ct = 0; ct < 4; ++ct) {
    const float* wrow = W + (size_t)(ct * 16 + r) * HEAD_DIM;  // B column e = W row e
    v8f acc = {};
    acc = wmma_f16(a0, fragB_f(wrow, 0, hf), acc);
    acc = wmma_f16(a1, fragB_f(wrow, 32, hf), acc);
    c[ct] = acc;
  }

  const size_t headBase = ((size_t)n * HEADS + h) * (size_t)S * HEAD_DIM;
#pragma unroll
  for (int ct = 0; ct < 4; ++ct) {
    const int e = ct * 16 + r;  // C layout: lane = output-dim column
#pragma unroll
    for (int i = 0; i < 8; ++i) {
      const int s = s0 + hf * 8 + i;  // C layout: VGPR i = row (token)
      if (!transpose)
        Out[headBase + (size_t)s * HEAD_DIM + e] = (_Float16)c[ct][i];
      else
        Out[headBase + (size_t)e * S + s] = (_Float16)c[ct][i];
    }
  }
}

// ---- kernel 2: flash attention, LDS-staged K/V with async double-buffering --
// Qp/Kp: [N,H,S,64] f16, Vt: [N,H,64,S] f16, Out: [N,S,EMBED] f16
// Block = 4 waves = 64 queries of one (n,h); 64-key chunks staged in LDS.
__global__ void __launch_bounds__(128) attn_kernel(const _Float16* __restrict__ Qp,
                                                   const _Float16* __restrict__ Kp,
                                                   const _Float16* __restrict__ Vt,
                                                   _Float16* __restrict__ Out,
                                                   int S) {
  __shared__ __align__(16) _Float16 lk[2][64 * HEAD_DIM];  // 64 keys x 64 dims
  __shared__ __align__(16) _Float16 lv[2][HEAD_DIM * 64];  // 64 dims x 64 keys (V^T rows)

  const int tid  = threadIdx.x;
  const int lane = tid & 31;
  const int wave = tid >> 5;
  const int r  = lane & 15;
  const int hf = lane >> 4;
  const int qt = blockIdx.x * 4 + wave;
  const int h  = blockIdx.y;
  const int n  = blockIdx.z;

  const size_t headBase = ((size_t)n * HEADS + h) * (size_t)S * HEAD_DIM;
  const _Float16* Qh = Qp + headBase;
  const _Float16* Kh = Kp + headBase;
  const _Float16* Vh = Vt + headBase;  // row d (stride S), col s

  // Q^T B-fragments (column = query r, K = head dim)
  const _Float16* qrow = Qh + (size_t)(qt * 16 + r) * HEAD_DIM;
  const v16h bq0 = fragB_h(qrow, 0, hf);
  const v16h bq1 = fragB_h(qrow, 32, hf);

  v8f o0 = {}, o1 = {}, o2 = {}, o3 = {};
  float m = -1e30f, l = 0.f;
  const float cexp = 1.4426950408889634f / 32.f;  // log2(e) / sqrt(EMBED)

  // --- stage chunk 0 ---
  {
    const _Float16* gk = Kh;  // rows kc..kc+63 are contiguous (8 KB)
#pragma unroll
    for (int i = 0; i < 4; ++i) {
      const int p = tid + i * 128;                 // 16B piece index, 512 total
      async_cp16(gk + p * 8, &lk[0][p * 8]);
    }
#pragma unroll
    for (int i = 0; i < 4; ++i) {
      const int p = tid + i * 128;
      const int d = p >> 3, j = p & 7;             // dim row, 16B piece in row
      async_cp16(Vh + (size_t)d * S + j * 8, &lv[0][d * 64 + j * 8]);
    }
  }

  for (int kc = 0; kc < S; kc += 64) {
    const int bi = (kc >> 6) & 1;
    wait_async0();       // my async copies into buf bi are done
    __syncthreads();     // everyone done reading buf 1-bi, all copies visible

    if (kc + 64 < S) {   // prefetch next chunk into the idle buffer
      const _Float16* gk = Kh + (size_t)(kc + 64) * HEAD_DIM;
#pragma unroll
      for (int i = 0; i < 4; ++i) {
        const int p = tid + i * 128;
        async_cp16(gk + p * 8, &lk[1 - bi][p * 8]);
      }
#pragma unroll
      for (int i = 0; i < 4; ++i) {
        const int p = tid + i * 128;
        const int d = p >> 3, j = p & 7;
        async_cp16(Vh + (size_t)d * S + (kc + 64) + j * 8, &lv[1 - bi][d * 64 + j * 8]);
      }
    }

    // ---- scores S^T for 64 keys: 4 tiles (M = key, N = query) ----
    v8f sc[4];
#pragma unroll
    for (int st = 0; st < 4; ++st) {
      const _Float16* krow = &lk[bi][(st * 16 + r) * HEAD_DIM];
      v8f s = {};
      s = wmma_f16(fragA_h(krow, 0, hf), bq0, s);
      s = wmma_f16(fragA_h(krow, 32, hf), bq1, s);
      sc[st] = s;
    }

    // ---- online softmax (per-lane query = r; lane^16 handles same query) ----
    float tm = sc[0][0];
#pragma unroll
    for (int st = 0; st < 4; ++st)
#pragma unroll
      for (int i = 0; i < 8; ++i) tm = fmaxf(tm, sc[st][i]);
    tm = fmaxf(tm, __shfl_xor(tm, 16, 32));
    const float mnew  = fmaxf(m, tm);
    const float alpha = exp2f((m - mnew) * cexp);

    v16h pw0, pw1;
    float psum = 0.f;
#pragma unroll
    for (int i = 0; i < 8; ++i) {
      float p = exp2f((sc[0][i] - mnew) * cexp); psum += p; pw0[i]     = (_Float16)p;
      p       = exp2f((sc[1][i] - mnew) * cexp); psum += p; pw0[i + 8] = (_Float16)p;
      p       = exp2f((sc[2][i] - mnew) * cexp); psum += p; pw1[i]     = (_Float16)p;
      p       = exp2f((sc[3][i] - mnew) * cexp); psum += p; pw1[i + 8] = (_Float16)p;
    }
    psum += __shfl_xor(psum, 16, 32);
    l = l * alpha + psum;

    const v16h bp0 = make_bp(pw0, hf);  // keys kc+0..31
    const v16h bp1 = make_bp(pw1, hf);  // keys kc+32..63

#pragma unroll
    for (int i = 0; i < 8; ++i) {
      o0[i] *= alpha; o1[i] *= alpha; o2[i] *= alpha; o3[i] *= alpha;
    }

    // ---- O^T += V^T(16 dims x 32 keys) x P^T, 4 dim tiles x 2 key halves ----
#pragma unroll
    for (int kk = 0; kk < 2; ++kk) {
      const v16h bp = kk ? bp1 : bp0;
      o0 = wmma_f16(fragA_h(&lv[bi][(0 * 16 + r) * 64 + kk * 32], 0, hf), bp, o0);
      o1 = wmma_f16(fragA_h(&lv[bi][(1 * 16 + r) * 64 + kk * 32], 0, hf), bp, o1);
      o2 = wmma_f16(fragA_h(&lv[bi][(2 * 16 + r) * 64 + kk * 32], 0, hf), bp, o2);
      o3 = wmma_f16(fragA_h(&lv[bi][(3 * 16 + r) * 64 + kk * 32], 0, hf), bp, o3);
    }
    m = mnew;
  }

  const float inv = 1.f / l;
  const int q = qt * 16 + r;  // O^T C layout: lane = query column
  _Float16* obase = Out + ((size_t)n * S + q) * EMBED + h * HEAD_DIM;
#pragma unroll
  for (int i = 0; i < 8; ++i) {
    const int d = hf * 8 + i;  // O^T rows = head dims
    obase[d]      = (_Float16)(o0[i] * inv);
    obase[d + 16] = (_Float16)(o1[i] * inv);
    obase[d + 32] = (_Float16)(o2[i] * inv);
    obase[d + 48] = (_Float16)(o3[i] * inv);
  }
}

// ---- kernel 3: output projection  out = A @ Wo^T + bo -----------------------
// A: [T, EMBED] f16, Wo: [EMBED, EMBED] f32, out: [T, EMBED] f32
__global__ void __launch_bounds__(128) outproj_kernel(const _Float16* __restrict__ A,
                                                      const float* __restrict__ Wo,
                                                      const float* __restrict__ bo,
                                                      float* __restrict__ Out, int T) {
  const int lane = threadIdx.x & 31;
  const int wave = threadIdx.x >> 5;
  const int r  = lane & 15;
  const int hf = lane >> 4;
  const int t0 = (blockIdx.x * 4 + wave) * 16;
  const int ebase = blockIdx.y * 64;

  const _Float16* arow = A + ((size_t)t0 + r) * EMBED;
  const float* wr0 = Wo + (size_t)(ebase + 0 + r) * EMBED;
  const float* wr1 = Wo + (size_t)(ebase + 16 + r) * EMBED;
  const float* wr2 = Wo + (size_t)(ebase + 32 + r) * EMBED;
  const float* wr3 = Wo + (size_t)(ebase + 48 + r) * EMBED;

  v8f c0 = {}, c1 = {}, c2 = {}, c3 = {};
  for (int k = 0; k < EMBED; k += 32) {
    const v16h af = fragA_h(arow, k, hf);
    c0 = wmma_f16(af, fragB_f(wr0, k, hf), c0);
    c1 = wmma_f16(af, fragB_f(wr1, k, hf), c1);
    c2 = wmma_f16(af, fragB_f(wr2, k, hf), c2);
    c3 = wmma_f16(af, fragB_f(wr3, k, hf), c3);
  }

  const float b0 = bo[ebase + 0 + r];
  const float b1 = bo[ebase + 16 + r];
  const float b2 = bo[ebase + 32 + r];
  const float b3 = bo[ebase + 48 + r];
#pragma unroll
  for (int i = 0; i < 8; ++i) {
    const size_t t = (size_t)t0 + hf * 8 + i;
    Out[t * EMBED + ebase + 0 + r]  = c0[i] + b0;
    Out[t * EMBED + ebase + 16 + r] = c1[i] + b1;
    Out[t * EMBED + ebase + 32 + r] = c2[i] + b2;
    Out[t * EMBED + ebase + 48 + r] = c3[i] + b3;
  }
}

extern "C" void kernel_launch(void* const* d_in, const int* in_sizes, int n_in,
                              void* d_out, int out_size, void* d_ws, size_t ws_size,
                              hipStream_t stream) {
  const float* values = (const float*)d_in[0];
  const float* keys   = (const float*)d_in[1];
  const float* query  = (const float*)d_in[2];
  const float* Wv = (const float*)d_in[3];
  const float* Wk = (const float*)d_in[4];
  const float* Wq = (const float*)d_in[5];
  const float* Wo = (const float*)d_in[6];
  const float* bo = (const float*)d_in[7];
  float* out = (float*)d_out;

  const int S = 2048;
  const int N = in_sizes[2] / (S * EMBED);
  const int T = N * S;

  // workspace carve-up (f16): Qp, Kp [N,H,S,64]; Vt [N,H,64,S]; Ao [N,S,EMBED]
  const size_t per = (size_t)N * HEADS * S * HEAD_DIM;
  _Float16* Qp = (_Float16*)d_ws;
  _Float16* Kp = Qp + per;
  _Float16* Vt = Kp + per;
  _Float16* Ao = Vt + per;

  const dim3 blk(128);  // 4 waves
  const dim3 gproj(T / 64, HEADS);
  proj_kernel<<<gproj, blk, 0, stream>>>(query,  Wq, Qp, S, 0);
  proj_kernel<<<gproj, blk, 0, stream>>>(keys,   Wk, Kp, S, 0);
  proj_kernel<<<gproj, blk, 0, stream>>>(values, Wv, Vt, S, 1);

  const dim3 gattn(S / 64, HEADS, N);
  attn_kernel<<<gattn, blk, 0, stream>>>(Qp, Kp, Vt, Ao, S);

  const dim3 gout(T / 64, EMBED / 64);
  outproj_kernel<<<gout, blk, 0, stream>>>(Ao, Wo, bo, out, T);
}
